// ExpertMLP_58179626991686
// MI455X (gfx1250) — matrix operand order repaired and checked by reading, not compiled, not run on hardware
//
#include <hip/hip_runtime.h>
#include <stdint.h>

typedef __attribute__((ext_vector_type(16))) __bf16 v16bf;
typedef __attribute__((ext_vector_type(8)))  float  v8f;

#define T_DIM 8192
#define D_DIM 1024
#define H_DIM 4096

#define BM 128
#define BN 128
#define BK 32
#define LDSROW 40                  // 32 + 8 pad bf16 -> 80B stride: conflict-free b128 frag loads
#define LDSTILE (BM * LDSROW)      // 5120 elems = 10240 B per tile buffer

union FragU { uint4 q[2]; v16bf v; };

// fp32 -> bf16 round-to-nearest-even
__device__ __forceinline__ uint16_t f2bf(float f) {
  uint32_t u = __builtin_bit_cast(uint32_t, f);
  uint32_t r = u + 0x7FFFu + ((u >> 16) & 1u);
  return (uint16_t)(r >> 16);
}

// CDNA5 async copy: global -> LDS, 16 bytes, tracked by ASYNCcnt
__device__ __forceinline__ void async_b128(uint32_t lds_addr, uint64_t gaddr) {
  asm volatile("global_load_async_to_lds_b128 %0, %1, off"
               :: "v"(lds_addr), "v"(gaddr) : "memory");
}

__device__ __forceinline__ void wait_async0() {
#if __has_builtin(__builtin_amdgcn_s_wait_asynccnt)
  __builtin_amdgcn_s_wait_asynccnt(0);
#else
  asm volatile("s_wait_asynccnt 0x0" ::: "memory");
#endif
}

// ---------------- fp32 -> bf16 bulk conversion (memory bound, one-shot) ---------
__global__ void __launch_bounds__(256)
cvt_f32_bf16(const float* __restrict__ src, uint16_t* __restrict__ dst, int n4) {
  int i = blockIdx.x * 256 + threadIdx.x;
  const int stride = gridDim.x * 256;
  for (; i < n4; i += stride) {
    float4 f = ((const float4*)src)[i];
    uint32_t lo = (uint32_t)f2bf(f.x) | ((uint32_t)f2bf(f.y) << 16);
    uint32_t hi = (uint32_t)f2bf(f.z) | ((uint32_t)f2bf(f.w) << 16);
    ((uint2*)dst)[i] = make_uint2(lo, hi);
  }
}

// ---------------- bf16 GEMM: out = act( A[M,K] @ B[N,K]^T ) --------------------
// 128x128 block tile, BK=32, 8 waves (4x2), async global->LDS double buffering.
template<bool RELU2, bool OUT_BF16>
__global__ void __launch_bounds__(256)
gemm_bf16_async(const uint16_t* __restrict__ A,   // [M,K] bf16 row-major
                const uint16_t* __restrict__ B,   // [N,K] bf16 row-major
                void* __restrict__ Outp,
                int M, int N, int K)
{
  __shared__ uint16_t smem[2 * 2 * LDSTILE];      // [buf][A|B][tile]

  const int tid  = threadIdx.x;
  const int lane = tid & 31;
  const int wave = tid >> 5;
  const int wm   = wave >> 1;                     // 0..3 : 32 output rows
  const int wn   = wave & 1;                      // 0..1 : 64 output cols
  const int bm   = blockIdx.y * BM;
  const int bn   = blockIdx.x * BN;

  v8f acc[2][4];
#pragma unroll
  for (int i = 0; i < 2; ++i)
#pragma unroll
    for (int j = 0; j < 4; ++j) acc[i][j] = v8f{};

  // Staging: 128 rows x 32 cols bf16 = 512 x 16B chunks; 256 threads x 2 chunks.
  // chunk id = tid + c*256 : row = id>>2, col = (id&3)*8
  uint64_t gA[2], gB[2];
  uint32_t lA[2][2], lB[2][2];                    // [chunk][buf] LDS byte addresses
#pragma unroll
  for (int c = 0; c < 2; ++c) {
    int id  = tid + c * 256;
    int row = id >> 2, col = (id & 3) * 8;
    gA[c] = (uint64_t)(uintptr_t)(A + (size_t)(bm + row) * K + col);
    gB[c] = (uint64_t)(uintptr_t)(B + (size_t)(bn + row) * K + col);
#pragma unroll
    for (int p = 0; p < 2; ++p) {
      lA[c][p] = (uint32_t)(uintptr_t)&smem[p * 2 * LDSTILE + row * LDSROW + col];
      lB[c][p] = (uint32_t)(uintptr_t)&smem[p * 2 * LDSTILE + LDSTILE + row * LDSROW + col];
    }
  }

  // Prime buffer 0 with the first K-slab.
#pragma unroll
  for (int c = 0; c < 2; ++c) {
    async_b128(lA[c][0], gA[c]);
    async_b128(lB[c][0], gB[c]);
  }

  int p = 0;
  for (int kt = 0; kt < K; kt += BK) {
    wait_async0();                 // my copies into buf p have landed in LDS
    __syncthreads();               // everyone's copies landed; everyone done reading buf p^1

    if (kt + BK < K) {             // kick off next slab into the other buffer
#pragma unroll
      for (int c = 0; c < 2; ++c) {
        gA[c] += BK * sizeof(uint16_t);
        gB[c] += BK * sizeof(uint16_t);
        async_b128(lA[c][p ^ 1], gA[c]);
        async_b128(lB[c][p ^ 1], gB[c]);
      }
    }

    const uint16_t* As = &smem[p * 2 * LDSTILE];
    const uint16_t* Bs = &smem[p * 2 * LDSTILE + LDSTILE];

    const int half = lane >> 4;
    const int r16  = lane & 15;
    FragU a[2], b[4];
    // A 16x32 layout: lane-half h holds K {8h..8h+7} in v0..3, {16+8h..+7} in v4..7
#pragma unroll
    for (int i = 0; i < 2; ++i) {
      int row = wm * 32 + i * 16 + r16;
      int ka  = half * 8;
      a[i].q[0] = *(const uint4*)&As[row * LDSROW + ka];
      a[i].q[1] = *(const uint4*)&As[row * LDSROW + ka + 16];
    }
    // B 32x16 layout: lane-half h holds K {8h..8h+7} in v0..3, {16+8h..+7} in v4..7
#pragma unroll
    for (int j = 0; j < 4; ++j) {
      int row = wn * 64 + j * 16 + r16;
      int kb  = half * 8;
      b[j].q[0] = *(const uint4*)&Bs[row * LDSROW + kb];
      b[j].q[1] = *(const uint4*)&Bs[row * LDSROW + kb + 16];
    }
#pragma unroll
    for (int i = 0; i < 2; ++i)
#pragma unroll
      for (int j = 0; j < 4; ++j)
        acc[i][j] = __builtin_amdgcn_wmma_f32_16x16x32_bf16(
            false, a[i].v, false, b[j].v, (short)0, acc[i][j], false, false);

    p ^= 1;
  }

  // Epilogue: D layout -> row = base + 8*(lane>=16) + v, col = base + (lane&15)
  const int half = lane >> 4;
  const int nc   = lane & 15;
#pragma unroll
  for (int i = 0; i < 2; ++i) {
    int mbase = bm + wm * 32 + i * 16 + half * 8;
#pragma unroll
    for (int j = 0; j < 4; ++j) {
      int col = bn + wn * 64 + j * 16 + nc;
#pragma unroll
      for (int v = 0; v < 8; ++v) {
        float val = acc[i][j][v];
        if constexpr (RELU2) { val = fmaxf(val, 0.0f); val *= val; }
        size_t idx = (size_t)(mbase + v) * (size_t)N + (size_t)col;
        if constexpr (OUT_BF16) ((uint16_t*)Outp)[idx] = f2bf(val);
        else                    ((float*)Outp)[idx]    = val;
      }
    }
  }
}

extern "C" void kernel_launch(void* const* d_in, const int* in_sizes, int n_in,
                              void* d_out, int out_size, void* d_ws, size_t ws_size,
                              hipStream_t stream) {
  const float* x      = (const float*)d_in[0];   // [T, DIM]
  const float* W_fc   = (const float*)d_in[1];   // [HID, DIM]
  const float* W_proj = (const float*)d_in[2];   // [DIM, HID]

  // Workspace layout (96 MB total):
  uint8_t* ws = (uint8_t*)d_ws;
  uint16_t* xb  = (uint16_t*)(ws);                                   // 16 MB
  uint16_t* wfb = (uint16_t*)(ws + (size_t)T_DIM * D_DIM * 2);       //  8 MB
  uint16_t* wpb = (uint16_t*)(ws + (size_t)T_DIM * D_DIM * 2
                                 + (size_t)H_DIM * D_DIM * 2);       //  8 MB
  uint16_t* H   = (uint16_t*)(ws + (size_t)T_DIM * D_DIM * 2
                                 + 2 * (size_t)H_DIM * D_DIM * 2);   // 64 MB

  dim3 blk(256, 1, 1);

  // One-shot fp32 -> bf16 conversions (memory bound)
  cvt_f32_bf16<<<dim3(2048), blk, 0, stream>>>(x,      xb,  T_DIM * D_DIM / 4);
  cvt_f32_bf16<<<dim3(1024), blk, 0, stream>>>(W_fc,   wfb, H_DIM * D_DIM / 4);
  cvt_f32_bf16<<<dim3(1024), blk, 0, stream>>>(W_proj, wpb, D_DIM * H_DIM / 4);

  // GEMM1: H = bf16( relu(x @ W_fc^T)^2 )   M=T, N=HID, K=DIM
  dim3 g1(H_DIM / BN, T_DIM / BM, 1);
  gemm_bf16_async<true, true><<<g1, blk, 0, stream>>>(
      xb, wfb, (void*)H, T_DIM, H_DIM, D_DIM);

  // GEMM2: out = H @ W_proj^T               M=T, N=DIM, K=HID
  dim3 g2(D_DIM / BN, T_DIM / BM, 1);
  gemm_bf16_async<false, false><<<g2, blk, 0, stream>>>(
      H, wpb, d_out, T_DIM, D_DIM, H_DIM);
}
